// Hgru3_43954695307924
// MI455X (gfx1250) — matrix operand order repaired and checked by reading, not compiled, not run on hardware
//
#include <hip/hip_runtime.h>
#include <hip/hip_bf16.h>
#include <math.h>

// ---------------------------------------------------------------------------
// CDNA5 (gfx1250) HGRU3/GLA forward.   B=2, T=1024, E=1024, H=16, D=64.
// GEMMs: V_WMMA_F32_16X16X32_BF16 (bf16 operands, f32 accum), LDS double-
// buffered so global loads for tile k+1 overlap WMMAs on tile k.
// ---------------------------------------------------------------------------

typedef __attribute__((ext_vector_type(16))) __bf16 v16bf;
typedef __attribute__((ext_vector_type(8)))  __bf16 v8bf;
typedef __attribute__((ext_vector_type(4)))  __bf16 v4bf;
typedef __attribute__((ext_vector_type(8)))  float  v8f;

enum { EP_NONE = 0, EP_SILU = 1, EP_LOGSIG = 2 };

__device__ __forceinline__ float sigmoidf_fast(float z) {
    return 1.0f / (1.0f + __expf(-z));
}
__device__ __forceinline__ float logsigmoidf_fast(float z) {  // stable
    return (z >= 0.0f) ? -log1pf(__expf(-z)) : (z - log1pf(__expf(z)));
}

// ---------------------------------------------------------------------------
// WMMA bf16 GEMM:  C[M,N] = A[M,K] (f32, row-major) x W[K,N] (f32, row-major)
// Block tile 128x128, K-step 32, 256 threads = 8 waves (2x4), each wave owns
// a 64x32 tile = 4x2 WMMA accumulators. Double-buffered LDS (2 x 16 KB).
// M % 128 == 0, K % 32 == 0. RAGGED=false requires N % 128 == 0.
// ---------------------------------------------------------------------------
template <int EPI, bool RAGGED>
__global__ __launch_bounds__(256) void gemm_bf16_wmma(
    const float* __restrict__ A, const float* __restrict__ W,
    float* __restrict__ C0, float* __restrict__ C1,
    const float* __restrict__ bias, int M, int N, int K) {
    __shared__ __align__(16) __bf16 As[2][128 * 32];   // [row][k]
    __shared__ __align__(32) __bf16 Bs[2][128 * 32];   // [n][k] (transposed)

    const int tid  = threadIdx.x;
    const int lane = tid & 31;
    const int wave = tid >> 5;
    const int wr   = wave >> 2;      // 0..1 : 64-row slab
    const int wc   = wave & 3;       // 0..3 : 32-col slab
    const int lnlo = lane & 15;
    const int kh   = lane >> 4;      // K-half selector (WMMA wave32 layout)

    const int m0blk = blockIdx.y * 128;
    const int n0blk = blockIdx.x * 128;

    // staging decomposition (4 float4 chunks per thread per matrix)
    int arow[4], acol[4], bkrow[4], bn[4];
#pragma unroll
    for (int i = 0; i < 4; ++i) {
        const int f = i * 256 + tid;       // 0..1023
        arow[i]  = f >> 3;                 // 0..127
        acol[i]  = (f & 7) << 2;           // 0,4,...,28
        bkrow[i] = f >> 5;                 // 0..31
        bn[i]    = (f & 31) << 2;          // 0,4,...,124
    }

    v8f acc[4][2];
#pragma unroll
    for (int mt = 0; mt < 4; ++mt)
#pragma unroll
        for (int nt = 0; nt < 2; ++nt) acc[mt][nt] = v8f{};

    float4 aReg[4], wReg[4];

    auto load_regs = [&](int k0) {
#pragma unroll
        for (int i = 0; i < 4; ++i)
            aReg[i] = *(const float4*)(
                &A[(size_t)(m0blk + arow[i]) * K + k0 + acol[i]]);
#pragma unroll
        for (int i = 0; i < 4; ++i) {
            const int gn = n0blk + bn[i];
            if (!RAGGED) {
                wReg[i] = *(const float4*)(&W[(size_t)(k0 + bkrow[i]) * N + gn]);
            } else {
                float4 w = make_float4(0.f, 0.f, 0.f, 0.f);
                const float* wp = &W[(size_t)(k0 + bkrow[i]) * N];
                if (gn + 3 < N) {
                    w = *(const float4*)(&wp[gn]);
                } else {
                    if (gn + 0 < N) w.x = wp[gn + 0];
                    if (gn + 1 < N) w.y = wp[gn + 1];
                    if (gn + 2 < N) w.z = wp[gn + 2];
                    if (gn + 3 < N) w.w = wp[gn + 3];
                }
                wReg[i] = w;
            }
        }
    };

    auto store_lds = [&](int buf) {
#pragma unroll
        for (int i = 0; i < 4; ++i) {
            v4bf c4;
            c4[0] = (__bf16)aReg[i].x; c4[1] = (__bf16)aReg[i].y;
            c4[2] = (__bf16)aReg[i].z; c4[3] = (__bf16)aReg[i].w;
            *(v4bf*)(&As[buf][arow[i] * 32 + acol[i]]) = c4;
        }
#pragma unroll
        for (int i = 0; i < 4; ++i) {
            __bf16* bp = &Bs[buf][bn[i] * 32 + bkrow[i]];
            bp[0 * 32] = (__bf16)wReg[i].x;
            bp[1 * 32] = (__bf16)wReg[i].y;
            bp[2 * 32] = (__bf16)wReg[i].z;
            bp[3 * 32] = (__bf16)wReg[i].w;
        }
    };

    // prologue: stage tile 0
    load_regs(0);
    store_lds(0);
    __syncthreads();

    int cur = 0;
    for (int k0 = 0; k0 < K; k0 += 32) {
        const bool hasNext = (k0 + 32 < K);
        if (hasNext) load_regs(k0 + 32);   // global loads in flight over WMMAs

        // ---- fragments per wave32 WMMA layouts -----------------------------
        union AF { v16bf v; v8bf h[2]; };
        v16bf afr[4];
#pragma unroll
        for (int mt = 0; mt < 4; ++mt) {
            const __bf16* ap = &As[cur][(wr * 64 + mt * 16 + lnlo) * 32];
            AF af;
            af.h[0] = *(const v8bf*)(ap + kh * 8);        // K = 8*kh .. +7
            af.h[1] = *(const v8bf*)(ap + 16 + kh * 8);   // K = 16+8*kh .. +7
            afr[mt] = af.v;
        }
        v16bf bfr[2];
#pragma unroll
        for (int nt = 0; nt < 2; ++nt) {
            const __bf16* bp = &Bs[cur][(wc * 32 + nt * 16 + lnlo) * 32];
            AF bf;
            bf.h[0] = *(const v8bf*)(bp + kh * 8);
            bf.h[1] = *(const v8bf*)(bp + 16 + kh * 8);
            bfr[nt] = bf.v;
        }

        // ---- 8 matrix ops per wave per K-step ------------------------------
#pragma unroll
        for (int mt = 0; mt < 4; ++mt)
#pragma unroll
            for (int nt = 0; nt < 2; ++nt)
                acc[mt][nt] = __builtin_amdgcn_wmma_f32_16x16x32_bf16(
                    false, afr[mt], false, bfr[nt], (short)0, acc[mt][nt],
                    false, false);

        if (hasNext) store_lds(cur ^ 1);   // cvt + LDS store for next tile
        __syncthreads();                   // one barrier per K-step
        cur ^= 1;
    }

    // ---- epilogue + store (C layout: VGPR r -> M = r + 8*kh, N = lnlo) -----
#pragma unroll
    for (int mt = 0; mt < 4; ++mt) {
#pragma unroll
        for (int nt = 0; nt < 2; ++nt) {
            const int col = n0blk + wc * 32 + nt * 16 + lnlo;
            if (RAGGED && col >= N) continue;
            const int rbase = m0blk + wr * 64 + mt * 16 + kh * 8;
#pragma unroll
            for (int r = 0; r < 8; ++r) {
                float val = acc[mt][nt][r];
                const size_t idx = (size_t)(rbase + r) * N + col;
                if (EPI == EP_NONE) {
                    C0[idx] = val;
                } else if (EPI == EP_SILU) {
                    C0[idx] = val * sigmoidf_fast(val);
                } else {  // EP_LOGSIG: log_f -> C0, k = -expm1(log_f) -> C1
                    float z  = val + bias[col];
                    float lg = logsigmoidf_fast(z);
                    C0[idx]  = lg;
                    C1[idx]  = -expm1f(lg);
                }
            }
        }
    }
}

// ---------------------------------------------------------------------------
// GLA scan: S_t = diag(exp(g_t)) S_{t-1} + k_t v_t^T ; o_t = q_t S_t.
// State columns independent -> B*H*D = 2048 independent length-T scans.
// grid = B*H*4; block = 512 = 16 waves; wave <-> one state column j
// (64 rows -> 2 per lane), wave32 shuffle reduction, LDS double-buffered
// so there is a single barrier per timestep.
// ---------------------------------------------------------------------------
__global__ __launch_bounds__(512) void gla_scan(
    const float* __restrict__ q, const float* __restrict__ kk,
    const float* __restrict__ vv, const float* __restrict__ lf,
    float* __restrict__ o) {
    const int E = 1024, D = 64, T = 1024;
    const int bid  = blockIdx.x;          // 0..127
    const int bh   = bid >> 2;            // b*16 + h
    const int colg = bid & 3;
    const int b    = bh >> 4;
    const int h    = bh & 15;
    const int tid  = threadIdx.x;
    const int lane = tid & 31;
    const int wave = tid >> 5;            // 0..15
    const int j    = colg * 16 + wave;    // state column 0..63

    __shared__ float qs[2][64], ks[2][64], es[2][64];

    float s0 = 0.0f, s1 = 0.0f;           // S[lane][j], S[lane+32][j]
    const int r0 = lane, r1 = lane + 32;
    const size_t rowbase = (size_t)b * T * E + (size_t)h * D;

    for (int t = 0; t < T; ++t) {
        const size_t base = rowbase + (size_t)t * E;
        const int pb = t & 1;
        if (tid < 64) {
            qs[pb][tid] = q[base + tid];
            ks[pb][tid] = kk[base + tid];
            es[pb][tid] = __expf(lf[base + tid]);
        }
        __syncthreads();
        const float vj = vv[base + j];
        s0 = fmaf(es[pb][r0], s0, ks[pb][r0] * vj);
        s1 = fmaf(es[pb][r1], s1, ks[pb][r1] * vj);
        float p = fmaf(qs[pb][r0], s0, qs[pb][r1] * s1);
#pragma unroll
        for (int off = 16; off; off >>= 1) p += __shfl_xor(p, off, 32);
        if (lane == 0) o[base + j] = p;
    }
}

// ---------------------------------------------------------------------------
// Fused sigmoid-gate + group(64)-RMSNorm + weight. One wave per group,
// 8 independent waves (groups) per block.
// ---------------------------------------------------------------------------
__global__ __launch_bounds__(256) void gate_group_norm(
    const float* __restrict__ o, const float* __restrict__ gate,
    const float* __restrict__ nw, float* __restrict__ on) {
    const int lane = threadIdx.x & 31;
    const int wave = threadIdx.x >> 5;
    const int g    = blockIdx.x * 8 + wave;   // bt*16 + h
    const int bt   = g >> 4;
    const int h    = g & 15;
    const size_t base = (size_t)bt * 1024 + h * 64;
    const int e0 = h * 64 + lane, e1 = e0 + 32;

    float v0 = o[base + lane]      * sigmoidf_fast(gate[base + lane]);
    float v1 = o[base + lane + 32] * sigmoidf_fast(gate[base + lane + 32]);
    float ss = v0 * v0 + v1 * v1;
#pragma unroll
    for (int off = 16; off; off >>= 1) ss += __shfl_xor(ss, off, 32);
    const float r = rsqrtf(ss * (1.0f / 64.0f) + 1e-5f);
    on[base + lane]      = v0 * r * nw[e0];
    on[base + lane + 32] = v1 * r * nw[e1];
}

// ---------------------------------------------------------------------------
extern "C" void kernel_launch(void* const* d_in, const int* in_sizes, int n_in,
                              void* d_out, int out_size, void* d_ws,
                              size_t ws_size, hipStream_t stream) {
    (void)in_sizes; (void)n_in; (void)out_size; (void)ws_size;
    const float* x     = (const float*)d_in[0];
    const float* Wq    = (const float*)d_in[1];
    const float* Wk    = (const float*)d_in[2];
    const float* Wv    = (const float*)d_in[3];
    const float* Wo    = (const float*)d_in[4];
    const float* delta = (const float*)d_in[5];
    const float* G1    = (const float*)d_in[6];
    const float* G2    = (const float*)d_in[7];
    const float* normw = (const float*)d_in[8];
    float* out = (float*)d_out;

    const int BT = 2048, E = 1024;
    const size_t BTE = (size_t)BT * E;            // 2,097,152 floats
    float* ws  = (float*)d_ws;
    float* q   = ws + 0 * BTE;
    float* kk  = ws + 1 * BTE;
    float* vv  = ws + 2 * BTE;
    float* lf  = ws + 3 * BTE;
    float* oo  = ws + 4 * BTE;
    float* on  = ws + 5 * BTE;
    float* gt  = ws + 6 * BTE;
    float* g1  = ws + 7 * BTE;                    // BT*64

    const dim3 blk(256);
    const dim3 gridE(E / 128, BT / 128);          // 8 x 16
    const dim3 gridG1(1, BT / 128);               // N = 64 -> ragged col tile

    // projections (fused activations)
    gemm_bf16_wmma<EP_SILU, false><<<gridE, blk, 0, stream>>>(
        x, Wq, q, nullptr, nullptr, BT, E, E);
    gemm_bf16_wmma<EP_LOGSIG, false><<<gridE, blk, 0, stream>>>(
        x, Wk, lf, kk, delta, BT, E, E);
    gemm_bf16_wmma<EP_NONE, false><<<gridE, blk, 0, stream>>>(
        x, Wv, vv, nullptr, nullptr, BT, E, E);

    // low-rank gate: (x @ G1) @ G2
    gemm_bf16_wmma<EP_NONE, true><<<gridG1, blk, 0, stream>>>(
        x, G1, g1, nullptr, nullptr, BT, 64, E);
    gemm_bf16_wmma<EP_NONE, false><<<gridE, blk, 0, stream>>>(
        g1, G2, gt, nullptr, nullptr, BT, E, 64);

    // sequential GLA recurrence, parallel over B*H*D columns
    gla_scan<<<128, 512, 0, stream>>>(q, kk, vv, lf, oo);

    // gate + group rmsnorm
    gate_group_norm<<<4096, 256, 0, stream>>>(oo, gt, normw, on);

    // output projection
    gemm_bf16_wmma<EP_NONE, false><<<gridE, blk, 0, stream>>>(
        on, Wo, out, nullptr, nullptr, BT, E, E);
}
